// GeoMoE_10797547782282
// MI455X (gfx1250) — matrix operand order repaired and checked by reading, not compiled, not run on hardware
//
#include <hip/hip_runtime.h>
#include <math.h>

// ---------------- problem constants (match reference) ----------------
#define DMODEL 1024
#define HDIM   768
#define NEXP   23
#define TOPK   3

typedef __attribute__((ext_vector_type(16))) _Float16 f16x16;
typedef __attribute__((ext_vector_type(8)))  float    f32x8;

__device__ __forceinline__ float gelu_exact(float v) {
    return 0.5f * v * (1.0f + erff(v * 0.70710678118654752f));
}

// ---------------------------------------------------------------------
// 1) fp32 -> fp16 convert (x)
// ---------------------------------------------------------------------
__global__ __launch_bounds__(256) void cvt_f32_f16_kernel(const float* __restrict__ in,
                                                          _Float16* __restrict__ out, int n4) {
    int i = blockIdx.x * 256 + threadIdx.x;
    if (i >= n4) return;
    float4 v = ((const float4*)in)[i];
    union { _Float16 h[4]; uint2 q; } pk;
    pk.h[0] = (_Float16)v.x; pk.h[1] = (_Float16)v.y;
    pk.h[2] = (_Float16)v.z; pk.h[3] = (_Float16)v.w;
    ((uint2*)out)[i] = pk.q;
}

// ---------------------------------------------------------------------
// 2) batched fp32 [Z][R][C] -> fp16 [Z][C][R] transpose-convert
// ---------------------------------------------------------------------
__global__ __launch_bounds__(256) void transpose_cvt_kernel(const float* __restrict__ in,
                                                            _Float16* __restrict__ out,
                                                            int R, int C) {
    __shared__ float tile[32][33];
    int tx = threadIdx.x & 31, ty = threadIdx.x >> 5;        // 32 x 8
    size_t zoff = (size_t)blockIdx.z * R * C;
    int c0 = blockIdx.x * 32, r0 = blockIdx.y * 32;
#pragma unroll
    for (int it = 0; it < 4; ++it) {
        int r = r0 + ty + it * 8, c = c0 + tx;
        float v = 0.f;
        if (r < R && c < C) v = in[zoff + (size_t)r * C + c];
        tile[ty + it * 8][tx] = v;
    }
    __syncthreads();
#pragma unroll
    for (int it = 0; it < 4; ++it) {
        int n = c0 + ty + it * 8;   // output row (C dim)
        int k = r0 + tx;            // output col (R dim)
        if (n < C && k < R) out[zoff + (size_t)n * R + k] = (_Float16)tile[tx][ty + it * 8];
    }
}

// ---------------------------------------------------------------------
// 3) gating: wave per token, fp32, sigmoid, top-3 (first-index tie-break)
// ---------------------------------------------------------------------
__global__ __launch_bounds__(128) void gate_kernel(const float* __restrict__ x,
                                                   const float* __restrict__ gw_mat,
                                                   const float* __restrict__ gb,
                                                   int T,
                                                   float* __restrict__ gwn,
                                                   int* __restrict__ topidx,
                                                   float* __restrict__ tw) {
    int lane = threadIdx.x & 31;
    int wid  = threadIdx.x >> 5;
    int t = blockIdx.x * 4 + wid;
    if (t >= T) return;
    const float* xr = x + (size_t)t * DMODEL;
    float myg = 0.f;                       // lane e (<23) holds sigmoid gate value
    for (int e = 0; e < NEXP; ++e) {
        float s = 0.f;
        for (int i = lane; i < DMODEL; i += 32) s += xr[i] * gw_mat[i * NEXP + e];
#pragma unroll
        for (int o = 16; o > 0; o >>= 1) s += __shfl_xor(s, o, 32);
        if (lane == e) myg = 1.f / (1.f + expf(-(s + gb[e])));
    }
    float sv = (lane < NEXP) ? myg : 0.f;
#pragma unroll
    for (int o = 16; o > 0; o >>= 1) sv += __shfl_xor(sv, o, 32);
    if (lane < NEXP) gwn[(size_t)t * NEXP + lane] = myg / sv;
    // serial top-3 via shuffle (all lanes run; lane0 writes). strict '>' => first index on ties
    int i1 = -1, i2 = -1, i3 = -1;
    float v1 = -1.f, v2 = -1.f, v3 = -1.f;
    for (int e = 0; e < NEXP; ++e) {
        float v = __shfl(myg, e, 32);
        if (v > v1)      { v3 = v2; i3 = i2; v2 = v1; i2 = i1; v1 = v; i1 = e; }
        else if (v > v2) { v3 = v2; i3 = i2; v2 = v;  i2 = e; }
        else if (v > v3) { v3 = v;  i3 = e; }
    }
    if (lane == 0) {
        float ts = v1 + v2 + v3;
        topidx[t * 3 + 0] = i1; topidx[t * 3 + 1] = i2; topidx[t * 3 + 2] = i3;
        tw[t * 3 + 0] = v1 / ts; tw[t * 3 + 1] = v2 / ts; tw[t * 3 + 2] = v3 / ts;
    }
}

// ---------------------------------------------------------------------
// 4) single-workgroup deterministic bucket build + aux loss
// ---------------------------------------------------------------------
__global__ __launch_bounds__(256) void bucket_aux_kernel(const int* __restrict__ topidx,
                                                         const float* __restrict__ gwn,
                                                         int T,
                                                         int* __restrict__ counts,
                                                         int* __restrict__ offsets,
                                                         int* __restrict__ rows,
                                                         int* __restrict__ pos,
                                                         float* __restrict__ auxOut) {
    __shared__ int   cidx[256][24];
    __shared__ int   totals[24];
    __shared__ int   offs[25];
    __shared__ float red[256];
    int tid = threadIdx.x;
    int total = T * TOPK;
    int chunk = (total + 255) / 256;
    int begin = tid * chunk;
    int end   = begin + chunk; if (end > total) end = total;
    for (int e = 0; e < 24; ++e) cidx[tid][e] = 0;
    for (int i = begin; i < end; ++i) cidx[tid][topidx[i]]++;
    __syncthreads();
    if (tid < NEXP) {               // exclusive prefix over chunks (deterministic)
        int run = 0;
        for (int c = 0; c < 256; ++c) { int v = cidx[c][tid]; cidx[c][tid] = run; run += v; }
        totals[tid] = run;
    }
    __syncthreads();
    if (tid == 0) {
        int r = 0;
        for (int e = 0; e < NEXP; ++e) { offs[e] = r; r += totals[e]; }
        offs[NEXP] = r;
        for (int e = 0; e < NEXP; ++e) { counts[e] = totals[e]; offsets[e] = offs[e]; }
    }
    __syncthreads();
    for (int i = begin; i < end; ++i) {
        int e = topidx[i];
        int slot = offs[e] + cidx[tid][e]++;
        rows[slot] = i / 3;          // token index
        pos[i]     = slot;           // bucket slot for (t,k)
    }
    // aux = sum_e mean_t(gwn[t,e]) * (E*counts_e/(K*T)), fixed-tree reduction
    float aux = 0.f;
    for (int e = 0; e < NEXP; ++e) {
        float p = 0.f;
        for (int t = tid; t < T; t += 256) p += gwn[(size_t)t * NEXP + e];
        red[tid] = p; __syncthreads();
        for (int s = 128; s > 0; s >>= 1) { if (tid < s) red[tid] += red[tid + s]; __syncthreads(); }
        if (tid == 0) {
            float P  = red[0] / (float)T;
            float Fv = (float)NEXP * (float)totals[e] / ((float)TOPK * (float)T);
            aux += P * Fv;
        }
        __syncthreads();
    }
    if (tid == 0) auxOut[0] = aux;
}

// ---------------------------------------------------------------------
// 5) grouped GEMM: C[M,N] = act(A[rows,K] @ BT[N,K]^T + bias)
//    block tile 128x64, 8 waves, wave tile 32x32 = 2x2 WMMA f16 frags.
//    Double-buffered LDS, register-staged global loads (1 barrier / step),
//    row gather hoisted out of the K loop.
//    Expert mode: counts!=null, blockIdx.z=expert, early-exit empty tiles.
// ---------------------------------------------------------------------
template <int GELU, typename OUTT>
__global__ __launch_bounds__(256) void gemm_tile_kernel(
    const _Float16* __restrict__ A, int lda,
    const _Float16* __restrict__ BT,          // [N][K] per expert
    const float* __restrict__ bias,
    OUTT* __restrict__ out, int ldo,
    int Kdim, int Ndim,
    const int* __restrict__ counts, const int* __restrict__ offsets, int Mtotal,
    const int* __restrict__ rowlist) {
    __shared__ _Float16 As[2][128 * 32];
    __shared__ _Float16 Bs[2][64 * 32];
    int tid = threadIdx.x;
    int cnt, off;
    if (counts) {
        int e = blockIdx.z;
        cnt = counts[e]; off = offsets[e];
        BT   += (size_t)e * Kdim * Ndim;
        bias += (size_t)e * Ndim;
    } else { cnt = Mtotal; off = 0; }
    int mbase = blockIdx.y * 128;
    if (mbase >= cnt) return;                 // uniform across block
    int nbase = blockIdx.x * 64;
    int lane = tid & 31, wave = tid >> 5;
    int wrow = wave & 3, wcol = wave >> 2;    // 4 x 2 wave grid

    // ---- hoisted per-thread staging sources (row gather done ONCE) ----
    // A staging: items it=0,1 -> item = tid + it*256; r = item>>2; c = (item&3)*8
    const _Float16* aSrc[2];
    bool aValid[2];
    int aR[2], aC[2];
#pragma unroll
    for (int it = 0; it < 2; ++it) {
        int item = tid + it * 256;
        aR[it] = item >> 2; aC[it] = (item & 3) << 3;
        int gm = mbase + aR[it];
        aValid[it] = (gm < cnt);
        int arow = 0;
        if (aValid[it]) arow = rowlist ? rowlist[off + gm] : (off + gm);
        aSrc[it] = A + (size_t)arow * lda + aC[it];
    }
    // B staging: r = tid>>2 (n), c = (tid&3)*8 (k)
    int bR = tid >> 2, bC = (tid & 3) << 3;
    const _Float16* bSrc = BT + (size_t)(nbase + bR) * Kdim + bC;

    // ---- hoisted fragment LDS offsets ----
    int ar = lane & 15;
    int ko = (lane >> 4) << 3;                 // A: K-half select
    int aOff[2];
#pragma unroll
    for (int mi = 0; mi < 2; ++mi) aOff[mi] = (wrow * 32 + mi * 16 + ar) * 32 + ko;
    int kk = (lane >> 4) << 3;                 // B: K-half select
    int bOff[2];
#pragma unroll
    for (int ni = 0; ni < 2; ++ni) bOff[ni] = (wcol * 32 + ni * 16 + (lane & 15)) * 32 + kk;

    f32x8 acc[2][2];
#pragma unroll
    for (int a = 0; a < 2; ++a)
#pragma unroll
        for (int b = 0; b < 2; ++b)
#pragma unroll
            for (int q = 0; q < 8; ++q) acc[a][b][q] = 0.f;

    const int nsteps = Kdim >> 5;             // K multiple of 32 (1024 / 768)
    const uint4 zero4 = make_uint4(0u, 0u, 0u, 0u);

    // prologue: load tile 0 -> buffer 0
    uint4 ra0 = aValid[0] ? *(const uint4*)(aSrc[0]) : zero4;
    uint4 ra1 = aValid[1] ? *(const uint4*)(aSrc[1]) : zero4;
    uint4 rb  = *(const uint4*)(bSrc);
    *(uint4*)(As[0] + aR[0] * 32 + aC[0]) = ra0;
    *(uint4*)(As[0] + aR[1] * 32 + aC[1]) = ra1;
    *(uint4*)(Bs[0] + bR * 32 + bC)       = rb;
    __syncthreads();

    for (int step = 0; step < nsteps; ++step) {
        int cur = step & 1, nxt = cur ^ 1;
        bool more = (step + 1) < nsteps;
        int k1 = (step + 1) << 5;
        // issue next tile's global loads early (overlap with WMMA compute)
        if (more) {
            ra0 = aValid[0] ? *(const uint4*)(aSrc[0] + k1) : zero4;
            ra1 = aValid[1] ? *(const uint4*)(aSrc[1] + k1) : zero4;
            rb  = *(const uint4*)(bSrc + k1);
            if (step + 2 < nsteps)
                __builtin_prefetch(bSrc + ((step + 2) << 5), 0, 1);
        }
        // compute from current buffer
        f16x16 afr[2], bfr[2];
#pragma unroll
        for (int mi = 0; mi < 2; ++mi) {
            union { f16x16 v; uint4 q[2]; } u;
            u.q[0] = *(const uint4*)(As[cur] + aOff[mi]);
            u.q[1] = *(const uint4*)(As[cur] + aOff[mi] + 16);
            afr[mi] = u.v;
        }
#pragma unroll
        for (int ni = 0; ni < 2; ++ni) {
            union { f16x16 v; uint4 q[2]; } u;
            u.q[0] = *(const uint4*)(Bs[cur] + bOff[ni]);
            u.q[1] = *(const uint4*)(Bs[cur] + bOff[ni] + 16);
            bfr[ni] = u.v;
        }
#pragma unroll
        for (int mi = 0; mi < 2; ++mi)
#pragma unroll
            for (int ni = 0; ni < 2; ++ni)
                acc[mi][ni] = __builtin_amdgcn_wmma_f32_16x16x32_f16(
                    false, afr[mi], false, bfr[ni], (short)0, acc[mi][ni], false, false);
        // stage next tile into the other buffer (no reader conflicts)
        if (more) {
            *(uint4*)(As[nxt] + aR[0] * 32 + aC[0]) = ra0;
            *(uint4*)(As[nxt] + aR[1] * 32 + aC[1]) = ra1;
            *(uint4*)(Bs[nxt] + bR * 32 + bC)       = rb;
        }
        __syncthreads();
    }

    // epilogue: C elem r -> M = 8*(lane>=16)+r, N = lane&15
    int nloc = lane & 15;
    int m8 = (lane >> 4) << 3;
#pragma unroll
    for (int mi = 0; mi < 2; ++mi)
#pragma unroll
        for (int ni = 0; ni < 2; ++ni) {
            int gn = nbase + wcol * 32 + ni * 16 + nloc;
            float bv = bias[gn];
#pragma unroll
            for (int r = 0; r < 8; ++r) {
                int gm = mbase + wrow * 32 + mi * 16 + m8 + r;
                if (gm < cnt) {
                    float v = acc[mi][ni][r] + bv;
                    if (GELU) v = gelu_exact(v);
                    out[(size_t)(off + gm) * ldo + gn] = (OUTT)v;
                }
            }
        }
}

// ---------------------------------------------------------------------
// 6) combine: out[t] = shared(already in out) + sum_k tw[t,k]*ebuf[pos[t,k]]
// ---------------------------------------------------------------------
__global__ __launch_bounds__(256) void combine_kernel(float* __restrict__ out,
                                                      const float* __restrict__ ebuf,
                                                      const int* __restrict__ pos,
                                                      const float* __restrict__ tw,
                                                      int T) {
    int idx = blockIdx.x * 256 + threadIdx.x;   // 4 floats each; 256 thr == 1 token
    int t = idx >> 8;
    int c = (idx & 255) << 2;
    if (t >= T) return;
    float4 v = *(const float4*)(out + (size_t)t * DMODEL + c);
#pragma unroll
    for (int k = 0; k < TOPK; ++k) {
        int slot = pos[t * 3 + k];
        float w  = tw[t * 3 + k];
        float4 e4 = *(const float4*)(ebuf + (size_t)slot * DMODEL + c);
        v.x += w * e4.x; v.y += w * e4.y; v.z += w * e4.z; v.w += w * e4.w;
    }
    *(float4*)(out + (size_t)t * DMODEL + c) = v;
}

// ---------------------------------------------------------------------
extern "C" void kernel_launch(void* const* d_in, const int* in_sizes, int n_in,
                              void* d_out, int out_size, void* d_ws, size_t ws_size,
                              hipStream_t stream) {
    const float* x      = (const float*)d_in[0];
    const float* gate_w = (const float*)d_in[1];
    const float* gate_b = (const float*)d_in[2];
    const float* w1     = (const float*)d_in[3];
    const float* b1     = (const float*)d_in[4];
    const float* w2     = (const float*)d_in[5];
    const float* b2     = (const float*)d_in[6];
    const float* ws1    = (const float*)d_in[7];
    const float* bs1    = (const float*)d_in[8];
    const float* ws2    = (const float*)d_in[9];
    const float* bs2    = (const float*)d_in[10];
    float* out = (float*)d_out;

    const int T = in_sizes[0] / DMODEL;        // 3136

    // workspace carve-up (256B aligned)
    char* wsb = (char*)d_ws;
    size_t cur = 0;
    auto alloc = [&](size_t bytes) -> void* {
        cur = (cur + 255) & ~(size_t)255;
        void* p = wsb + cur; cur += bytes; return p;
    };
    _Float16* xb    = (_Float16*)alloc((size_t)T * DMODEL * 2);
    _Float16* w1bT  = (_Float16*)alloc((size_t)NEXP * HDIM * DMODEL * 2);   // [e][h][d]
    _Float16* w2bT  = (_Float16*)alloc((size_t)NEXP * DMODEL * HDIM * 2);   // [e][d][h]
    _Float16* ws1bT = (_Float16*)alloc((size_t)HDIM * DMODEL * 2);
    _Float16* ws2bT = (_Float16*)alloc((size_t)DMODEL * HDIM * 2);
    float*    gwn   = (float*)alloc((size_t)T * NEXP * 4);
    int*      tidx  = (int*)alloc((size_t)T * TOPK * 4);
    float*    tww   = (float*)alloc((size_t)T * TOPK * 4);
    int*      cnts  = (int*)alloc(32 * 4);
    int*      offs  = (int*)alloc(32 * 4);
    int*      rows  = (int*)alloc((size_t)T * TOPK * 4);
    int*      pos   = (int*)alloc((size_t)T * TOPK * 4);
    _Float16* hbuf  = (_Float16*)alloc((size_t)T * TOPK * HDIM * 2);
    _Float16* hsbuf = (_Float16*)alloc((size_t)T * HDIM * 2);
    float*    ebuf  = (float*)alloc((size_t)T * TOPK * DMODEL * 4);

    dim3 blk(256);

    // 1) convert x
    {
        int n4 = T * DMODEL / 4;
        cvt_f32_f16_kernel<<<(n4 + 255) / 256, blk, 0, stream>>>(x, xb, n4);
    }
    // 2) transpose-convert weights
    transpose_cvt_kernel<<<dim3(HDIM / 32, DMODEL / 32, NEXP), blk, 0, stream>>>(w1, w1bT, DMODEL, HDIM);
    transpose_cvt_kernel<<<dim3(DMODEL / 32, HDIM / 32, NEXP), blk, 0, stream>>>(w2, w2bT, HDIM, DMODEL);
    transpose_cvt_kernel<<<dim3(HDIM / 32, DMODEL / 32, 1),   blk, 0, stream>>>(ws1, ws1bT, DMODEL, HDIM);
    transpose_cvt_kernel<<<dim3(DMODEL / 32, HDIM / 32, 1),   blk, 0, stream>>>(ws2, ws2bT, HDIM, DMODEL);
    // 3) gating (fp32)
    gate_kernel<<<(T + 3) / 4, dim3(128), 0, stream>>>(x, gate_w, gate_b, T, gwn, tidx, tww);
    // 4) buckets + aux (aux written to d_out[T*d])
    bucket_aux_kernel<<<1, blk, 0, stream>>>(tidx, gwn, T, cnts, offs, rows, pos,
                                             out + (size_t)T * DMODEL);
    // 5) GEMMs
    int gy = (T + 127) / 128;
    // experts: H = gelu(x[rows] @ w1 + b1)
    gemm_tile_kernel<1, _Float16><<<dim3(HDIM / 64, gy, NEXP), blk, 0, stream>>>(
        xb, DMODEL, w1bT, b1, hbuf, HDIM, DMODEL, HDIM, cnts, offs, 0, rows);
    // shared: Hs = gelu(x @ ws1 + bs1)
    gemm_tile_kernel<1, _Float16><<<dim3(HDIM / 64, gy, 1), blk, 0, stream>>>(
        xb, DMODEL, ws1bT, bs1, hsbuf, HDIM, DMODEL, HDIM, nullptr, nullptr, T, nullptr);
    // experts: ebuf = H @ w2 + b2
    gemm_tile_kernel<0, float><<<dim3(DMODEL / 64, gy, NEXP), blk, 0, stream>>>(
        hbuf, HDIM, w2bT, b2, ebuf, DMODEL, HDIM, DMODEL, cnts, offs, 0, nullptr);
    // shared: out = Hs @ ws2 + bs2   (written directly to d_out)
    gemm_tile_kernel<0, float><<<dim3(DMODEL / 64, gy, 1), blk, 0, stream>>>(
        hsbuf, HDIM, ws2bT, bs2, out, DMODEL, HDIM, DMODEL, nullptr, nullptr, T, nullptr);
    // 6) combine (deterministic fixed-order weighted sum)
    combine_kernel<<<T, blk, 0, stream>>>(out, ebuf, pos, tww, T);
}
